// Prediction_27496380629687
// MI455X (gfx1250) — matrix operand order repaired and checked by reading, not compiled, not run on hardware
//
#include <hip/hip_runtime.h>
#include <math.h>

// ---------------------------------------------------------------------------
// CDNA5 (gfx1250) wave32 WMMA types
// ---------------------------------------------------------------------------
typedef __bf16 bf16;
typedef __attribute__((ext_vector_type(16))) __bf16 bf16x16;
typedef __attribute__((ext_vector_type(8)))  __bf16 bf16x8;
typedef __attribute__((ext_vector_type(8)))  float  f32x8;
typedef unsigned int u32;
typedef __attribute__((ext_vector_type(4))) unsigned int u32x4;
typedef __attribute__((ext_vector_type(8))) int i32x8;
typedef __attribute__((ext_vector_type(4))) int i32x4;

#define NEGV (-1e12f)

// Tensor Data Mover availability (this toolchain: 6-arg builtin form)
#if defined(__AMDGCN__) && defined(__has_builtin)
#if __has_builtin(__builtin_amdgcn_tensor_load_to_lds) && \
    __has_builtin(__builtin_amdgcn_s_wait_tensorcnt)
#define USE_TDM 1
#endif
#endif
#ifndef USE_TDM
#define USE_TDM 0
#endif

__device__ __forceinline__ f32x8 wmma_bf16(bf16x16 a, bf16x16 b, f32x8 c) {
  // v_wmma_f32_16x16x32_bf16  (D = A(16x32) * B(32x16) + C, fp32 accum)
  return __builtin_amdgcn_wmma_f32_16x16x32_bf16(false, a, false, b,
                                                 (short)0, c, false, false);
}

// A fragment (16x32 bf16), row-major source, stride lda (elements).
// lane l: m = l&15, h = l>>4 ; element j: k = (j<8 ? h*8+j : 16+h*8+(j-8))
__device__ __forceinline__ bf16x16 load_a(const bf16* __restrict__ A, int lda,
                                          int m0, int k0, int lane) {
  int m = m0 + (lane & 15);
  int h = lane >> 4;
  const bf16* p = A + (size_t)m * lda + k0 + h * 8;
  bf16x8 lo = *(const bf16x8*)(p);
  bf16x8 hi = *(const bf16x8*)(p + 16);
  bf16x16 r;
#pragma unroll
  for (int j = 0; j < 8; ++j) { r[j] = lo[j]; r[j + 8] = hi[j]; }
  return r;
}

// B fragment (32x16 bf16) from B^T stored row-major [N][K] (ldk = K elems).
// lane l: n = l&15, h = l>>4 ; element j: k = (j<8 ? h*8+j : 16+h*8+(j-8))
__device__ __forceinline__ bf16x16 load_bt(const bf16* __restrict__ Bt, int ldk,
                                           int n0, int k0, int lane) {
  int n = n0 + (lane & 15);
  int h = lane >> 4;
  const bf16* p = Bt + (size_t)n * ldk + k0 + h * 8;
  bf16x8 lo = *(const bf16x8*)(p);
  bf16x8 hi = *(const bf16x8*)(p + 16);
  bf16x16 r;
#pragma unroll
  for (int j = 0; j < 8; ++j) { r[j] = lo[j]; r[j + 8] = hi[j]; }
  return r;
}

__device__ __forceinline__ float sigm(float x) { return 1.0f / (1.0f + __expf(-x)); }

#if USE_TDM
// Build a 2D tensor-DMA descriptor per CDNA5 ISA ch8 and issue TENSOR_LOAD_TO_LDS.
//   group0: [1:0]count=1 | lds_addr | global_addr[56:0] | type=2
//   group1: data_size[17:16] | tensor_dim0 (79:48) | tensor_dim1 (111:80)
//           | tile_dim0 (127:112) | tile_dim1 (143:128) | dim0_stride (207:160)
__device__ __forceinline__ void tdm_load_2d(u32 lds_off, const void* gptr,
                                            u32 tile_d0, u32 tile_d1,
                                            u32 tensor_d0, u32 tensor_d1,
                                            u32 stride_d0, u32 dsz_code) {
  unsigned long long ga = (unsigned long long)gptr;
  u32x4 g0;
  g0[0] = 1u;                                            // count=1 (valid user D#)
  g0[1] = lds_off;                                       // lds_addr (bytes)
  g0[2] = (u32)(ga & 0xffffffffu);                       // global_addr[31:0]
  g0[3] = (u32)((ga >> 32) & 0x01ffffffu) | (2u << 30);  // addr[56:32] | type=2
  i32x8 g1;
  g1[0] = (int)(dsz_code << 16);                         // wg_mask=0 | data_size
  g1[1] = (int)((tensor_d0 & 0xffffu) << 16);            // tensor_dim0[15:0]
  g1[2] = (int)(((tensor_d0 >> 16) & 0xffffu) |
                ((tensor_d1 & 0xffffu) << 16));          // dim0[31:16]|dim1[15:0]
  g1[3] = (int)(((tensor_d1 >> 16) & 0xffffu) |
                (tile_d0 << 16));                        // dim1[31:16]|tile_dim0
  g1[4] = (int)(tile_d1 & 0xffffu);                      // tile_dim1, tile_dim2=0
  g1[5] = (int)stride_d0;                                // dim0_stride[31:0]
  g1[6] = 0;                                             // stride[47:32]=0
  g1[7] = 0;
  i32x4 z4 = {0, 0, 0, 0};                               // 2D: groups 2/3 unused
  i32x8 z8 = {0, 0, 0, 0, 0, 0, 0, 0};
  __builtin_amdgcn_tensor_load_to_lds(g0, g1, z4, z4, z8, 0);
}
#endif

// ---------------------------------------------------------------------------
// fp32 -> bf16 conversion (activations, row-major)
// ---------------------------------------------------------------------------
__global__ void k_cvt(const float* __restrict__ src, bf16* __restrict__ dst, int n) {
  int i = blockIdx.x * blockDim.x + threadIdx.x;
  if (i < n) dst[i] = (bf16)src[i];
}

// fp32 [K][512] -> bf16 transposed [512][K]  (weights; N is always 512)
__global__ void k_cvt_t512(const float* __restrict__ src, bf16* __restrict__ dst, int K) {
  int i = blockIdx.x * blockDim.x + threadIdx.x;
  if (i >= K * 512) return;
  int k = i >> 9;
  int n = i & 511;
  dst[(size_t)n * K + k] = (bf16)src[i];
}

// ---------------------------------------------------------------------------
// Gating: current_node = has_left ? tanh(g_r)*sig(t_r) : tanh(g_l)*sig(t_l)
// 4 waves/block, block tile 16x64, full fusion of 4 GEMMs + gate epilogue.
// ---------------------------------------------------------------------------
__global__ void k_gating(const bf16* __restrict__ ce, const bf16* __restrict__ lc,
                         const bf16* __restrict__ WclT, const bf16* __restrict__ WclgT,
                         const bf16* __restrict__ WcrT, const bf16* __restrict__ WcrgT,
                         const float* __restrict__ bcl, const float* __restrict__ bclg,
                         const float* __restrict__ bcr, const float* __restrict__ bcrg,
                         const int* __restrict__ has_left,
                         float* __restrict__ node_out, bf16* __restrict__ node_bf) {
  const int lane = threadIdx.x & 31;
  const int wave = threadIdx.x >> 5;
  const int m0 = blockIdx.y * 16;
  const int n0 = blockIdx.x * 64 + wave * 16;
  f32x8 aGL = {}, aTL = {}, aGR = {}, aTR = {};
  for (int k0 = 0; k0 < 512; k0 += 32) {
    bf16x16 aC = load_a(ce, 512, m0, k0, lane);
    bf16x16 aL = load_a(lc, 512, m0, k0, lane);
    aGL = wmma_bf16(aC, load_bt(WclT,  512,  n0, k0,       lane), aGL);
    aTL = wmma_bf16(aC, load_bt(WclgT, 512,  n0, k0,       lane), aTL);
    aGR = wmma_bf16(aL, load_bt(WcrT,  1024, n0, k0,       lane), aGR);
    aGR = wmma_bf16(aC, load_bt(WcrT,  1024, n0, k0 + 512, lane), aGR);
    aTR = wmma_bf16(aL, load_bt(WcrgT, 1024, n0, k0,       lane), aTR);
    aTR = wmma_bf16(aC, load_bt(WcrgT, 1024, n0, k0 + 512, lane), aTR);
  }
  const int h = lane >> 4;
  const int n = n0 + (lane & 15);
#pragma unroll
  for (int r = 0; r < 8; ++r) {
    int m = m0 + h * 8 + r;
    float v;
    if (has_left[m])
      v = tanhf(aGR[r] + bcr[n]) * sigm(aTR[r] + bcrg[n]);
    else
      v = tanhf(aGL[r] + bcl[n]) * sigm(aTL[r] + bclg[n]);
    node_out[m * 512 + n] = v;
    node_bf[m * 512 + n]  = (bf16)v;
  }
}

// ---------------------------------------------------------------------------
// Generic C = A(bf16)[M,K] * Bt(bf16)[N,K]^T + bias, fp32 out. 16x64 per block.
// ---------------------------------------------------------------------------
__global__ void k_gemm_bias(const bf16* __restrict__ A, int K,
                            const bf16* __restrict__ Bt,
                            const float* __restrict__ bias,
                            float* __restrict__ C, int N) {
  const int lane = threadIdx.x & 31;
  const int wave = threadIdx.x >> 5;
  const int m0 = blockIdx.y * 16;
  const int n0 = blockIdx.x * 64 + wave * 16;
  f32x8 acc = {};
  for (int k0 = 0; k0 < K; k0 += 32)
    acc = wmma_bf16(load_a(A, K, m0, k0, lane), load_bt(Bt, K, n0, k0, lane), acc);
  const int h = lane >> 4;
  const int n = n0 + (lane & 15);
#pragma unroll
  for (int r = 0; r < 8; ++r) {
    int m = m0 + h * 8 + r;
    C[(size_t)m * N + n] = acc[r] + bias[n];
  }
}

// ---------------------------------------------------------------------------
// Attention energies (dominant GEMM, 34.4 GFLOP after Q-split):
//   E[b,s] = sum_n tanh( (enc[s,b,:] @ Wa2)[n] + Q[b,n] ) * ws[n] + bs
// 8 waves = 2 row-groups x 4 col-groups; 32 rows x 512 cols per block.
// enc tile (32x32 fp32) staged in LDS: TDM double-buffered async when
// available, cooperative load otherwise. fp32->bf16 cvt at fragment build.
// ---------------------------------------------------------------------------
__global__ void k_energies(const float* __restrict__ enc,
                           const bf16* __restrict__ Wa2T,
                           const float* __restrict__ Qb,
                           const float* __restrict__ wsv,
                           const float* __restrict__ bs,
                           const int* __restrict__ seq_mask,
                           float* __restrict__ energies) {
  __shared__ __align__(16) float ldsA[2][32 * 32];
  __shared__ float rowsum[32];
  const int t = threadIdx.x;           // 0..255
  const int lane = t & 31;
  const int wave = t >> 5;             // 0..7
  const int mi = wave & 1;             // row group (16 rows)
  const int ni = wave >> 1;            // col group (128 cols)
  const int r0 = blockIdx.x * 32;
  const int h = lane >> 4;
  const int nl = lane & 15;

  if (t < 32) rowsum[t] = 0.0f;
  f32x8 zero = {};
  f32x8 acc[8];
#pragma unroll
  for (int i = 0; i < 8; ++i) acc[i] = zero;

#if USE_TDM
  if (wave == 0)
    tdm_load_2d((u32)(uintptr_t)&ldsA[0][0], enc + (size_t)r0 * 512,
                /*tile*/32, 32, /*tensor*/512, 65536, /*stride*/512, /*f32*/2);
#endif
  __syncthreads();

  int cur = 0;
  for (int k0 = 0; k0 < 512; k0 += 32, cur ^= 1) {
#if USE_TDM
    if (wave == 0) __builtin_amdgcn_s_wait_tensorcnt(0);  // tile 'cur' landed
    __syncthreads();
    if (wave == 0 && k0 + 32 < 512)                       // prefetch next tile
      tdm_load_2d((u32)(uintptr_t)&ldsA[cur ^ 1][0],
                  enc + (size_t)r0 * 512 + (k0 + 32), 32, 32, 512, 65536, 512, 2);
#else
    {
      int row = t >> 3;                // 32 rows, 8 threads/row, 4 floats each
      int col = (t & 7) * 4;
      *(float4*)&ldsA[cur][row * 32 + col] =
          *(const float4*)(enc + (size_t)(r0 + row) * 512 + k0 + col);
    }
    __syncthreads();
#endif
    // A fragment shared by the 4 col-group waves (fp32 LDS -> bf16)
    bf16x16 a;
    {
      const float* pa = &ldsA[cur][(mi * 16 + nl) * 32];
#pragma unroll
      for (int j = 0; j < 8; ++j) {
        a[j]     = (bf16)pa[h * 8 + j];
        a[j + 8] = (bf16)pa[16 + h * 8 + j];
      }
    }
#pragma unroll
    for (int nf = 0; nf < 8; ++nf) {
      bf16x16 b = load_bt(Wa2T, 512, ni * 128 + nf * 16, k0, lane);
      acc[nf] = wmma_bf16(a, b, acc[nf]);
    }
    __syncthreads();
  }

  // fused epilogue: +Q[b], tanh, *ws, per-row reduction via ds_add_f32
#pragma unroll
  for (int r = 0; r < 8; ++r) {
    int m_loc = mi * 16 + h * 8 + r;
    int row = r0 + m_loc;              // row = s*256 + b
    int bb = row & 255;
    float partial = 0.0f;
#pragma unroll
    for (int nf = 0; nf < 8; ++nf) {
      int n = ni * 128 + nf * 16 + nl;
      partial += tanhf(acc[nf][r] + Qb[bb * 512 + n]) * wsv[n];
    }
    atomicAdd(&rowsum[m_loc], partial);
  }
  __syncthreads();
  if (t < 32) {
    int row = r0 + t;
    int s = row >> 8, bb = row & 255;
    energies[bb * 256 + s] = seq_mask[bb * 256 + s] ? NEGV : (rowsum[t] + bs[0]);
  }
}

// ---------------------------------------------------------------------------
// Softmax over S=256 per batch row (in place energies -> attn)
// ---------------------------------------------------------------------------
__global__ void k_softmax(float* __restrict__ att) {
  __shared__ float red[256];
  const int b = blockIdx.x, s = threadIdx.x;
  float e = att[b * 256 + s];
  red[s] = e;
  __syncthreads();
  for (int off = 128; off > 0; off >>= 1) {
    if (s < off) red[s] = fmaxf(red[s], red[s + off]);
    __syncthreads();
  }
  float mx = red[0];
  __syncthreads();
  float ex = __expf(e - mx);
  red[s] = ex;
  __syncthreads();
  for (int off = 128; off > 0; off >>= 1) {
    if (s < off) red[s] += red[s + off];
    __syncthreads();
  }
  att[b * 256 + s] = ex / red[0];
}

// ---------------------------------------------------------------------------
// context[b,h] = sum_s attn[b,s] * enc[s,b,h] ; also build leaf_input (bf16)
// ---------------------------------------------------------------------------
__global__ void k_context(const float* __restrict__ att,
                          const float* __restrict__ enc,
                          float* __restrict__ ctx_out,
                          const bf16* __restrict__ node_bf,
                          bf16* __restrict__ leaf_bf) {
  const int b = blockIdx.x;
  const int hh = threadIdx.x;          // 0..511
  float sum = 0.0f;
  for (int s = 0; s < 256; ++s)
    sum += att[b * 256 + s] * enc[((size_t)s * 256 + b) * 512 + hh];
  ctx_out[b * 512 + hh] = sum;
  leaf_bf[(size_t)b * 1024 + hh]       = node_bf[b * 512 + hh];
  leaf_bf[(size_t)b * 1024 + 512 + hh] = (bf16)sum;
}

// ---------------------------------------------------------------------------
// emb_w = concat(broadcast(emb_weight), num_pades) -> d_out + bf16 copy
// ---------------------------------------------------------------------------
__global__ void k_embw(const float* __restrict__ emb_weight,
                       const float* __restrict__ num_pades,
                       float* __restrict__ embw_out, bf16* __restrict__ embw_bf) {
  int i = blockIdx.x * blockDim.x + threadIdx.x;  // over 256*12*512
  if (i >= 256 * 12 * 512) return;
  int hh = i & 511;
  int bm = i >> 9;
  int m = bm % 12, b = bm / 12;
  float v = (m < 4) ? emb_weight[m * 512 + hh]
                    : num_pades[((size_t)b * 8 + (m - 4)) * 512 + hh];
  embw_out[i] = v;
  embw_bf[i]  = (bf16)v;
}

// ---------------------------------------------------------------------------
// num_score[b,m] = sum_n tanh( (emb_w[b,m,:] @ Ws2)[n] + P[b,n] ) * ws_score[n]
// same fused structure; rows = b*12+m (M=3072), 2x4 wave grid, 32 rows/block
// ---------------------------------------------------------------------------
__global__ void k_numscore(const bf16* __restrict__ embw_bf,
                           const bf16* __restrict__ Ws2T,
                           const float* __restrict__ Pb,
                           const float* __restrict__ wss,
                           const int* __restrict__ mask_nums,
                           float* __restrict__ num_out) {
  __shared__ float rowsum[32];
  const int t = threadIdx.x;
  const int lane = t & 31;
  const int wave = t >> 5;
  const int mi = wave & 1;
  const int ni = wave >> 1;
  const int r0 = blockIdx.x * 32;
  const int h = lane >> 4;
  const int nl = lane & 15;
  if (t < 32) rowsum[t] = 0.0f;
  __syncthreads();
  f32x8 zero = {};
  f32x8 acc[8];
#pragma unroll
  for (int i = 0; i < 8; ++i) acc[i] = zero;
  for (int k0 = 0; k0 < 512; k0 += 32) {
    bf16x16 a = load_a(embw_bf, 512, r0 + mi * 16, k0, lane);
#pragma unroll
    for (int nf = 0; nf < 8; ++nf) {
      bf16x16 b = load_bt(Ws2T, 512, ni * 128 + nf * 16, k0, lane);
      acc[nf] = wmma_bf16(a, b, acc[nf]);
    }
  }
#pragma unroll
  for (int r = 0; r < 8; ++r) {
    int m_loc = mi * 16 + h * 8 + r;
    int row = r0 + m_loc;              // row = b*12 + m
    int bb = row / 12;
    float partial = 0.0f;
#pragma unroll
    for (int nf = 0; nf < 8; ++nf) {
      int n = ni * 128 + nf * 16 + nl;
      partial += tanhf(acc[nf][r] + Pb[bb * 512 + n]) * wss[n];
    }
    atomicAdd(&rowsum[m_loc], partial);
  }
  __syncthreads();
  if (t < 32) {
    int row = r0 + t;
    num_out[row] = mask_nums[row] ? NEGV : rowsum[t];
  }
}

// ---------------------------------------------------------------------------
// Tiny heads: p_leaf (softmax over 2) and op (6 logits), K=1024 dot products
// ---------------------------------------------------------------------------
__global__ void k_heads(const float* __restrict__ node, const float* __restrict__ ctx,
                        const float* __restrict__ W_isleaf, const float* __restrict__ b_isleaf,
                        const float* __restrict__ W_ops, const float* __restrict__ b_ops,
                        float* __restrict__ pleaf_out, float* __restrict__ op_out) {
  __shared__ float red[8][256];
  const int b = blockIdx.x, t = threadIdx.x;
  float acc[8];
#pragma unroll
  for (int j = 0; j < 8; ++j) acc[j] = 0.0f;
  for (int k = t; k < 1024; k += 256) {
    float x = (k < 512) ? node[b * 512 + k] : ctx[b * 512 + (k - 512)];
    acc[0] += x * W_isleaf[k * 2 + 0];
    acc[1] += x * W_isleaf[k * 2 + 1];
#pragma unroll
    for (int j = 0; j < 6; ++j) acc[2 + j] += x * W_ops[k * 6 + j];
  }
#pragma unroll
  for (int j = 0; j < 8; ++j) red[j][t] = acc[j];
  __syncthreads();
  for (int off = 128; off > 0; off >>= 1) {
    if (t < off) {
#pragma unroll
      for (int j = 0; j < 8; ++j) red[j][t] += red[j][t + off];
    }
    __syncthreads();
  }
  if (t == 0) {
    float l0 = red[0][0] + b_isleaf[0];
    float l1 = red[1][0] + b_isleaf[1];
    float mx = fmaxf(l0, l1);
    float e0 = __expf(l0 - mx), e1 = __expf(l1 - mx);
    float s = e0 + e1;
    pleaf_out[b * 2 + 0] = e0 / s;
    pleaf_out[b * 2 + 1] = e1 / s;
#pragma unroll
    for (int j = 0; j < 6; ++j) op_out[b * 6 + j] = red[2 + j][0] + b_ops[j];
  }
}

// ---------------------------------------------------------------------------
// host orchestration
// ---------------------------------------------------------------------------
extern "C" void kernel_launch(void* const* d_in, const int* in_sizes, int n_in,
                              void* d_out, int out_size, void* d_ws, size_t ws_size,
                              hipStream_t stream) {
  const float* current_emb = (const float*)d_in[0];
  const float* left_child  = (const float*)d_in[1];
  const int*   has_left    = (const int*)  d_in[2];
  const float* enc         = (const float*)d_in[3];
  const float* num_pades   = (const float*)d_in[4];
  const int*   seq_mask    = (const int*)  d_in[5];
  const int*   mask_nums   = (const int*)  d_in[6];
  const float* emb_weight  = (const float*)d_in[7];
  const float* W_cl   = (const float*)d_in[8];   const float* b_cl   = (const float*)d_in[9];
  const float* W_clg  = (const float*)d_in[10];  const float* b_clg  = (const float*)d_in[11];
  const float* W_cr   = (const float*)d_in[12];  const float* b_cr   = (const float*)d_in[13];
  const float* W_crg  = (const float*)d_in[14];  const float* b_crg  = (const float*)d_in[15];
  const float* W_isleaf = (const float*)d_in[16]; const float* b_isleaf = (const float*)d_in[17];
  const float* W_ops  = (const float*)d_in[18];  const float* b_ops  = (const float*)d_in[19];
  const float* Wa_attn = (const float*)d_in[20]; const float* ba_attn = (const float*)d_in[21];
  const float* ws_attn = (const float*)d_in[22]; const float* bs_attn = (const float*)d_in[23];
  const float* Wa_score = (const float*)d_in[24]; const float* ba_score = (const float*)d_in[25];
  const float* ws_score = (const float*)d_in[26];

  // output layout (flat fp32, return order)
  float* out       = (float*)d_out;
  float* out_pleaf = out;               // [256,2]
  float* out_numsc = out + 512;         // [256,12]
  float* out_op    = out + 3584;        // [256,6]
  float* out_node  = out + 5120;        // [256,512]
  float* out_ctx   = out + 136192;      // [256,1,512]
  float* out_embw  = out + 267264;      // [256,12,512]

  // scratch carve-out (256B aligned)
  char* w = (char*)d_ws;
  auto carve = [&](size_t bytes) -> void* {
    void* p = (void*)w;
    w += (bytes + 255) & ~(size_t)255;
    return p;
  };
  bf16* ce_bf   = (bf16*)carve(256 * 512 * 2);
  bf16* lc_bf   = (bf16*)carve(256 * 512 * 2);
  bf16* node_bf = (bf16*)carve(256 * 512 * 2);
  bf16* leaf_bf = (bf16*)carve(256 * 1024 * 2);
  bf16* embw_bf = (bf16*)carve((size_t)3072 * 512 * 2);
  bf16* WclT    = (bf16*)carve(512 * 512 * 2);
  bf16* WclgT   = (bf16*)carve(512 * 512 * 2);
  bf16* WcrT    = (bf16*)carve(512 * 1024 * 2);
  bf16* WcrgT   = (bf16*)carve(512 * 1024 * 2);
  bf16* WaQT    = (bf16*)carve(512 * 512 * 2);
  bf16* Wa2T    = (bf16*)carve(512 * 512 * 2);
  bf16* WpT     = (bf16*)carve(512 * 1024 * 2);
  bf16* Ws2T    = (bf16*)carve(512 * 512 * 2);
  float* Qbuf   = (float*)carve(256 * 512 * 4);
  float* Pbuf   = (float*)carve(256 * 512 * 4);
  float* Ebuf   = (float*)carve(256 * 256 * 4);   // energies -> attn (in place)

  // 1) activation conversions
  k_cvt<<<512, 256, 0, stream>>>(current_emb, ce_bf, 256 * 512);
  k_cvt<<<512, 256, 0, stream>>>(left_child,  lc_bf, 256 * 512);

  // 2) weight transposes (fp32 [K][512] -> bf16 [512][K])
  k_cvt_t512<<<1024, 256, 0, stream>>>(W_cl,  WclT,  512);
  k_cvt_t512<<<1024, 256, 0, stream>>>(W_clg, WclgT, 512);
  k_cvt_t512<<<2048, 256, 0, stream>>>(W_cr,  WcrT,  1024);
  k_cvt_t512<<<2048, 256, 0, stream>>>(W_crg, WcrgT, 1024);
  k_cvt_t512<<<1024, 256, 0, stream>>>(Wa_attn,             WaQT, 512);    // rows 0..511
  k_cvt_t512<<<1024, 256, 0, stream>>>(Wa_attn + 512 * 512, Wa2T, 512);    // rows 512..1023
  k_cvt_t512<<<2048, 256, 0, stream>>>(Wa_score,              WpT,  1024); // rows 0..1023
  k_cvt_t512<<<1024, 256, 0, stream>>>(Wa_score + 1024 * 512, Ws2T, 512);  // rows 1024..1535

  // 3) gating -> current_node (output 4) + bf16 copy
  k_gating<<<dim3(8, 16), 128, 0, stream>>>(ce_bf, lc_bf, WclT, WclgT, WcrT, WcrgT,
                                            b_cl, b_clg, b_cr, b_crg, has_left,
                                            out_node, node_bf);

  // 4) Q = node @ Wa_attn[:H] + ba_attn
  k_gemm_bias<<<dim3(8, 16), 128, 0, stream>>>(node_bf, 512, WaQT, ba_attn, Qbuf, 512);

  // 5) dominant GEMM: energies [B,S] (mask + bias fused, TDM-staged A)
  k_energies<<<2048, 256, 0, stream>>>(enc, Wa2T, Qbuf, ws_attn, bs_attn, seq_mask, Ebuf);

  // 6) softmax over S, in place
  k_softmax<<<256, 256, 0, stream>>>(Ebuf);

  // 7) context (output 5) + leaf_input bf16
  k_context<<<256, 512, 0, stream>>>(Ebuf, enc, out_ctx, node_bf, leaf_bf);

  // 8) emb_w (output 6) + bf16 copy
  k_embw<<<6144, 256, 0, stream>>>(emb_weight, num_pades, out_embw, embw_bf);

  // 9) P = leaf @ Wa_score[:2H] + ba_score
  k_gemm_bias<<<dim3(8, 16), 128, 0, stream>>>(leaf_bf, 1024, WpT, ba_score, Pbuf, 512);

  // 10) num_score (output 2)
  k_numscore<<<96, 256, 0, stream>>>(embw_bf, Ws2T, Pbuf, ws_score, mask_nums, out_numsc);

  // 11) p_leaf (output 1) + op (output 3)
  k_heads<<<256, 256, 0, stream>>>(out_node, out_ctx, W_isleaf, b_isleaf,
                                   W_ops, b_ops, out_pleaf, out_op);
}
